// MultiHeadAttention_8581344658047
// MI455X (gfx1250) — matrix operand order repaired and checked by reading, not compiled, not run on hardware
//
#include <hip/hip_runtime.h>
#include <hip/hip_bf16.h>

// ---------------- types / helpers ----------------
typedef _Float16 h8   __attribute__((ext_vector_type(8)));
typedef _Float16 h16  __attribute__((ext_vector_type(16)));
typedef float    v8f  __attribute__((ext_vector_type(8)));
typedef int      v4i  __attribute__((vector_size(16)));

#define D_MODEL 1024
#define NUM_HEADS 16
#define D_K 64
#define BATCH 4
#define SEQ 2048
#define M_ROWS (BATCH * SEQ)          // 8192

__device__ __forceinline__ h8 ld8(const _Float16* p)  { return *(const h8*)p; }
__device__ __forceinline__ h16 ld16(const _Float16* p){ return *(const h16*)p; }

__device__ __forceinline__ h16 cat16(h8 a, h8 b) {
  h16 r;
#pragma unroll
  for (int i = 0; i < 8; ++i) { r[i] = a[i]; r[i + 8] = b[i]; }
  return r;
}

// convert 8 contiguous fp32 -> 8 f16
__device__ __forceinline__ h8 cvt8(const float* p) {
  const float4 a = ((const float4*)p)[0];
  const float4 b = ((const float4*)p)[1];
  h8 r;
  r[0] = (_Float16)a.x; r[1] = (_Float16)a.y; r[2] = (_Float16)a.z; r[3] = (_Float16)a.w;
  r[4] = (_Float16)b.x; r[5] = (_Float16)b.y; r[6] = (_Float16)b.z; r[7] = (_Float16)b.w;
  return r;
}

__device__ __forceinline__ v8f wmma16(h16 a, h16 b, v8f c) {
  return __builtin_amdgcn_wmma_f32_16x16x32_f16(false, a, false, b, (short)0, c,
                                                false, false);
}

// ---- async global->LDS (16B per lane), double-buffer staging ----
#if __has_builtin(__builtin_amdgcn_global_load_async_to_lds_b128)
#define HAS_ASYNC_LDS 1
__device__ __forceinline__ void async_cp16(void* lds, const void* g) {
  __builtin_amdgcn_global_load_async_to_lds_b128(
      (__attribute__((address_space(1))) v4i*)g,
      (__attribute__((address_space(3))) v4i*)lds, 0, 0);
}
#else
#define HAS_ASYNC_LDS 0
__device__ __forceinline__ void async_cp16(void* lds, const void* g) {
  *(h8*)lds = *(const h8*)g;   // synchronous fallback
}
#endif

#if __has_builtin(__builtin_amdgcn_s_wait_asynccnt)
#define WAIT_ASYNC(n) __builtin_amdgcn_s_wait_asynccnt(n)
#else
#define WAIT_ASYNC(n) asm volatile("s_wait_asynccnt " #n ::: "memory")
#endif

// ---------------- kernel 1: weight transpose + f16 convert ----------------
__global__ void k_wt(const float* __restrict__ Wq, const float* __restrict__ Wk,
                     const float* __restrict__ Wv, const float* __restrict__ Wo,
                     _Float16* __restrict__ WT) {
  const float* W = (blockIdx.z == 0) ? Wq : (blockIdx.z == 1) ? Wk
                 : (blockIdx.z == 2) ? Wv : Wo;
  _Float16* Dst = WT + (size_t)blockIdx.z * (D_MODEL * D_MODEL);
  __shared__ float t[32][33];
  const int tx = threadIdx.x, ty = threadIdx.y;
  const int n = blockIdx.x * 32 + tx;
  const int k0 = blockIdx.y * 32;
#pragma unroll
  for (int j = 0; j < 32; j += 8)
    t[ty + j][tx] = W[(size_t)(k0 + ty + j) * D_MODEL + n];
  __syncthreads();
  const int n2 = blockIdx.x * 32 + ty;
#pragma unroll
  for (int j = 0; j < 32; j += 8)
    Dst[(size_t)(n2 + j) * D_MODEL + k0 + tx] = (_Float16)t[tx][ty + j];
}

// ---------------- kernel 2: fused QKV projection ----------------
// Block = one (which, 16-row m-tile, 512-col half). Wave owns 16x64 (4 n-tiles),
// A-tile (16x32 fp32) staged via async global->LDS, double-buffered.
__global__ void k_qkv(const float* __restrict__ Xq, const float* __restrict__ Xk,
                      const float* __restrict__ Xv, const _Float16* __restrict__ WT,
                      const float* __restrict__ bq, const float* __restrict__ bk,
                      const float* __restrict__ bv,
                      _Float16* __restrict__ Q16, _Float16* __restrict__ K16,
                      _Float16* __restrict__ VT16) {
  const int wave = threadIdx.x >> 5;
  const int lane = threadIdx.x & 31;
  const int q = lane & 15;
  const int h = lane >> 4;

  const int bid = blockIdx.x;            // [0, 3*512*2)
  const int which = bid >> 10;
  const int rem = bid & 1023;
  const int mt = rem >> 1;
  const int side = rem & 1;
  const int m0 = mt * 16;
  const int nbase = side * 512 + wave * 64;

  const float* X = (which == 0) ? Xq : (which == 1) ? Xk : Xv;
  const float* bias = (which == 0) ? bq : (which == 1) ? bk : bv;
  const _Float16* W = WT + (size_t)which * (D_MODEL * D_MODEL);

  __shared__ __align__(128) float xa[2][16][32];   // 4 KB double-buffered A tile

  const int t = threadIdx.x;
  const int srow = t >> 3, sch = (t & 7) * 4;      // 128 threads x b128

  v8f acc[4];
#pragma unroll
  for (int i = 0; i < 4; ++i) acc[i] = (v8f){};

  if (t < 128)
    async_cp16(&xa[0][srow][sch], X + (size_t)(m0 + srow) * D_MODEL + sch);

  for (int it = 0; it < 32; ++it) {
    const int kk = it * 32;
    const int cur = it & 1;
    if (it + 1 < 32 && t < 128)
      async_cp16(&xa[(it + 1) & 1][srow][sch],
                 X + (size_t)(m0 + srow) * D_MODEL + kk + 32 + sch);
    WAIT_ASYNC(1);
    __syncthreads();
    const h16 a = cat16(cvt8(&xa[cur][q][16 * h]), cvt8(&xa[cur][q][16 * h + 8]));
#pragma unroll
    for (int i = 0; i < 4; ++i) {
      const int n0 = nbase + i * 16;
      __builtin_prefetch(W + (size_t)(n0 + q) * D_MODEL + kk + 64, 0, 0);
      acc[i] = wmma16(a, ld16(W + (size_t)(n0 + q) * D_MODEL + kk + 16 * h), acc[i]);
    }
    __syncthreads();
  }

  const int b = m0 >> 11;
  const int sl0 = m0 & (SEQ - 1);
#pragma unroll
  for (int i = 0; i < 4; ++i) {
    const int n = nbase + i * 16 + q;
    const int head = n >> 6;
    const int d = n & 63;
    const float bn = bias[n];
    if (which == 2) {
      h8 r;
#pragma unroll
      for (int v = 0; v < 8; ++v) r[v] = (_Float16)(acc[i][v] + bn);
      *(h8*)(VT16 + ((size_t)(b * NUM_HEADS + head) * D_K + d) * SEQ + sl0 + 8 * h) = r;
    } else {
      _Float16* O = (which == 0) ? Q16 : K16;
#pragma unroll
      for (int v = 0; v < 8; ++v) {
        const int row = m0 + v + 8 * h;
        const int sl = row & (SEQ - 1);
        O[((size_t)(b * NUM_HEADS + head) * SEQ + sl) * D_K + d] = (_Float16)(acc[i][v] + bn);
      }
    }
  }
}

// ---------------- kernel 3: flash attention ----------------
// Block = (b*h, 8 query tiles); K/V 32-key tiles staged once per block via
// async global->LDS (8x reuse across waves), double-buffered.
__global__ void k_attn(const _Float16* __restrict__ Q16,
                       const _Float16* __restrict__ K16,
                       const _Float16* __restrict__ VT16,
                       _Float16* __restrict__ CTX) {
  const int wave = threadIdx.x >> 5;
  const int lane = threadIdx.x & 31;
  const int q = lane & 15;
  const int h = lane >> 4;

  const int bh = blockIdx.x >> 4;                   // b*16+head
  const int qt = (blockIdx.x & 15) * 8 + wave;      // query tile

  const _Float16* Qb = Q16 + ((size_t)bh * SEQ + qt * 16) * D_K;
  const _Float16* Kb = K16 + (size_t)bh * SEQ * D_K;
  const _Float16* Vb = VT16 + (size_t)bh * D_K * SEQ;

  const h16 qf0 = cat16(ld8(Qb + q * D_K + 0  + 8 * h), ld8(Qb + q * D_K + 16 + 8 * h));
  const h16 qf1 = cat16(ld8(Qb + q * D_K + 32 + 8 * h), ld8(Qb + q * D_K + 48 + 8 * h));

  __shared__ __align__(128) _Float16 ks[2][32][64];   // 8 KB
  __shared__ __align__(128) _Float16 vs[2][64][32];   // 8 KB
  __shared__ __align__(128) _Float16 pl[8][16][32];   // 8 KB

  const int t = threadIdx.x;
  const int krow = t >> 3, kch = (t & 7) * 8;   // 256 x b128 for K tile
  const int vrow = t >> 2, vch = (t & 3) * 8;   // 256 x b128 for V tile

  const float scale = 0.125f;
  float m_run = -1e30f, l_run = 0.0f;
  v8f O0 = {}, O1 = {}, O2 = {}, O3 = {};

  async_cp16(&ks[0][krow][kch], Kb + (size_t)krow * D_K + kch);
  async_cp16(&vs[0][vrow][vch], Vb + (size_t)vrow * SEQ + vch);

  for (int it = 0; it < SEQ / 32; ++it) {
    const int kt = it * 32;
    const int cur = it & 1;
    if (it + 1 < SEQ / 32) {
      const int nb = (it + 1) & 1;
      async_cp16(&ks[nb][krow][kch], Kb + (size_t)(kt + 32 + krow) * D_K + kch);
      async_cp16(&vs[nb][vrow][vch], Vb + (size_t)vrow * SEQ + kt + 32 + vch);
    }
    WAIT_ASYNC(2);
    __syncthreads();

    // ---- scores^T = K . Q^T from LDS tiles
    const _Float16* kr0 = &ks[cur][q][0];
    const _Float16* kr1 = &ks[cur][16 + q][0];
    h16 a;
    v8f s0 = {}, s1 = {};
    a = cat16(ld8(kr0 + 8 * h), ld8(kr0 + 16 + 8 * h));       s0 = wmma16(a, qf0, s0);
    a = cat16(ld8(kr0 + 32 + 8 * h), ld8(kr0 + 48 + 8 * h));  s0 = wmma16(a, qf1, s0);
    a = cat16(ld8(kr1 + 8 * h), ld8(kr1 + 16 + 8 * h));       s1 = wmma16(a, qf0, s1);
    a = cat16(ld8(kr1 + 32 + 8 * h), ld8(kr1 + 48 + 8 * h));  s1 = wmma16(a, qf1, s1);

    // ---- running softmax statistics (per column q)
    float mx = -1e30f;
#pragma unroll
    for (int v = 0; v < 8; ++v) {
      s0[v] *= scale; s1[v] *= scale;
      mx = fmaxf(mx, fmaxf(s0[v], s1[v]));
    }
    mx = fmaxf(mx, __shfl_xor(mx, 16, 32));
    const float m_new = fmaxf(m_run, mx);
    const float factor = __expf(m_run - m_new);
    float psum = 0.0f;
    h8 p0, p1;
#pragma unroll
    for (int v = 0; v < 8; ++v) {
      const float e0 = __expf(s0[v] - m_new);
      const float e1 = __expf(s1[v] - m_new);
      p0[v] = (_Float16)e0; p1[v] = (_Float16)e1;
      psum += e0 + e1;
    }
    psum += __shfl_xor(psum, 16, 32);
    l_run = l_run * factor + psum;
    m_run = m_new;

    // ---- re-layout P via LDS into B-fragment form
    *(h8*)&pl[wave][q][ 0 + 8 * h] = p0;
    *(h8*)&pl[wave][q][16 + 8 * h] = p1;
    asm volatile("s_wait_dscnt 0x0" ::: "memory");
    const h16 pf = cat16(ld8(&pl[wave][q][8 * h]), ld8(&pl[wave][q][16 + 8 * h]));

    // ---- rescale ctx^T and accumulate V^T.P from LDS V tile
#pragma unroll
    for (int v = 0; v < 8; ++v) { O0[v] *= factor; O1[v] *= factor; O2[v] *= factor; O3[v] *= factor; }
    const _Float16* vr0 = &vs[cur][ 0 + q][0];
    const _Float16* vr1 = &vs[cur][16 + q][0];
    const _Float16* vr2 = &vs[cur][32 + q][0];
    const _Float16* vr3 = &vs[cur][48 + q][0];
    O0 = wmma16(cat16(ld8(vr0 + 8 * h), ld8(vr0 + 16 + 8 * h)), pf, O0);
    O1 = wmma16(cat16(ld8(vr1 + 8 * h), ld8(vr1 + 16 + 8 * h)), pf, O1);
    O2 = wmma16(cat16(ld8(vr2 + 8 * h), ld8(vr2 + 16 + 8 * h)), pf, O2);
    O3 = wmma16(cat16(ld8(vr3 + 8 * h), ld8(vr3 + 16 + 8 * h)), pf, O3);
    __syncthreads();
  }

  // ---- normalize and store ctx (f16, [B*S][1024])
  const float inv = 1.0f / l_run;
  const int b = bh >> 4, head = bh & 15;
  const size_t row = (size_t)b * SEQ + qt * 16 + q;
  _Float16* crow = CTX + row * D_MODEL + head * D_K;
  h8 r;
#pragma unroll
  for (int v = 0; v < 8; ++v) r[v] = (_Float16)(O0[v] * inv);
  *(h8*)(crow + 0  + 8 * h) = r;
#pragma unroll
  for (int v = 0; v < 8; ++v) r[v] = (_Float16)(O1[v] * inv);
  *(h8*)(crow + 16 + 8 * h) = r;
#pragma unroll
  for (int v = 0; v < 8; ++v) r[v] = (_Float16)(O2[v] * inv);
  *(h8*)(crow + 32 + 8 * h) = r;
#pragma unroll
  for (int v = 0; v < 8; ++v) r[v] = (_Float16)(O3[v] * inv);
  *(h8*)(crow + 48 + 8 * h) = r;
}

// ---------------- kernel 4: output projection + residual + LayerNorm ----------------
__global__ void k_outln(const _Float16* __restrict__ CTX,
                        const _Float16* __restrict__ WoT,
                        const float* __restrict__ bo,
                        const float* __restrict__ resid,
                        const float* __restrict__ gamma,
                        const float* __restrict__ beta,
                        float* __restrict__ out) {
  const int wave = threadIdx.x >> 5;
  const int lane = threadIdx.x & 31;
  const int q = lane & 15;
  const int h = lane >> 4;
  const int m0 = blockIdx.x * 16;

  __shared__ float Ot[16][D_MODEL];                  // 64 KB
  __shared__ __align__(128) _Float16 as_[2][16][32]; // 2 KB staged A tile

  const int t = threadIdx.x;
  const int srow = t >> 2, sch = (t & 3) * 8;        // 64 threads x b128

  v8f acc[8];
#pragma unroll
  for (int i = 0; i < 8; ++i) acc[i] = (v8f){};

  if (t < 64)
    async_cp16(&as_[0][srow][sch], CTX + (size_t)(m0 + srow) * D_MODEL + sch);

  for (int it = 0; it < 32; ++it) {
    const int kk = it * 32;
    const int cur = it & 1;
    if (it + 1 < 32 && t < 64)
      async_cp16(&as_[(it + 1) & 1][srow][sch],
                 CTX + (size_t)(m0 + srow) * D_MODEL + kk + 32 + sch);
    WAIT_ASYNC(1);
    __syncthreads();
    const h16 a = cat16(ld8(&as_[cur][q][16 * h]), ld8(&as_[cur][q][16 * h + 8]));
#pragma unroll
    for (int i = 0; i < 8; ++i) {
      const int n0 = (wave * 8 + i) * 16;
      __builtin_prefetch(WoT + (size_t)(n0 + q) * D_MODEL + kk + 64, 0, 0);
      acc[i] = wmma16(a, ld16(WoT + (size_t)(n0 + q) * D_MODEL + kk + 16 * h), acc[i]);
    }
    __syncthreads();
  }
#pragma unroll
  for (int i = 0; i < 8; ++i) {
    const int n = (wave * 8 + i) * 16 + q;
    const float bn = bo[n];
#pragma unroll
    for (int v = 0; v < 8; ++v) Ot[v + 8 * h][n] = acc[i][v] + bn;
  }
  __syncthreads();

#pragma unroll
  for (int rr = 0; rr < 2; ++rr) {
    const int r = 2 * wave + rr;
    const size_t grow = (size_t)(m0 + r);
    float sum = 0.0f, sq = 0.0f;
#pragma unroll 8
    for (int i = 0; i < 32; ++i) {
      const int j = lane + 32 * i;
      const float x = Ot[r][j] + resid[grow * D_MODEL + j];
      Ot[r][j] = x;
      sum += x; sq += x * x;
    }
#pragma unroll
    for (int off = 16; off > 0; off >>= 1) {
      sum += __shfl_xor(sum, off, 32);
      sq  += __shfl_xor(sq, off, 32);
    }
    const float mu = sum * (1.0f / D_MODEL);
    const float var = sq * (1.0f / D_MODEL) - mu * mu;
    const float rs = rsqrtf(var + 1e-5f);
#pragma unroll 8
    for (int i = 0; i < 32; ++i) {
      const int j = lane + 32 * i;
      out[grow * D_MODEL + j] = (Ot[r][j] - mu) * rs * gamma[j] + beta[j];
    }
  }
}

// ---------------- host-side launcher ----------------
extern "C" void kernel_launch(void* const* d_in, const int* in_sizes, int n_in,
                              void* d_out, int out_size, void* d_ws, size_t ws_size,
                              hipStream_t stream) {
  const float* Xq = (const float*)d_in[0];
  const float* Xk = (const float*)d_in[1];
  const float* Xv = (const float*)d_in[2];
  const float* Wq = (const float*)d_in[3];
  const float* bq = (const float*)d_in[4];
  const float* Wk = (const float*)d_in[5];
  const float* bk = (const float*)d_in[6];
  const float* Wv = (const float*)d_in[7];
  const float* bv = (const float*)d_in[8];
  const float* Wo = (const float*)d_in[9];
  const float* bo = (const float*)d_in[10];
  const float* gamma = (const float*)d_in[11];
  const float* beta  = (const float*)d_in[12];
  float* out = (float*)d_out;

  // workspace layout (72 MB total)
  char* ws = (char*)d_ws;
  _Float16* WT   = (_Float16*)(ws);                          //  8 MB
  _Float16* Q16  = (_Float16*)(ws + (size_t)8  * 1048576);   // 16 MB
  _Float16* K16  = (_Float16*)(ws + (size_t)24 * 1048576);   // 16 MB
  _Float16* VT16 = (_Float16*)(ws + (size_t)40 * 1048576);   // 16 MB
  _Float16* CTX  = (_Float16*)(ws + (size_t)56 * 1048576);   // 16 MB

  k_wt<<<dim3(32, 32, 4), dim3(32, 8), 0, stream>>>(Wq, Wk, Wv, Wo, WT);
  k_qkv<<<dim3(3 * 512 * 2), dim3(256), 0, stream>>>(
      Xq, Xk, Xv, WT, bq, bk, bv, Q16, K16, VT16);
  k_attn<<<dim3(BATCH * NUM_HEADS * (SEQ / 16) / 8), dim3(256), 0, stream>>>(
      Q16, K16, VT16, CTX);
  k_outln<<<dim3(M_ROWS / 16), dim3(256), 0, stream>>>(
      CTX, WT + (size_t)3 * D_MODEL * D_MODEL, bo, Xq, gamma, beta, out);
}
